// DiTBlock_58737972740502
// MI455X (gfx1250) — 8-bit matrix operand order repaired and checked by reading, compiled, not run on hardware
//
#include <hip/hip_runtime.h>
#include <hip/hip_bf16.h>
#include <cstdint>
#include <cstddef>


typedef __attribute__((ext_vector_type(8))) int v8i;
typedef int gv4i __attribute__((vector_size(16)));

#if defined(__AMDGCN__) && __has_builtin(__builtin_amdgcn_global_load_async_to_lds_b128)
#define USE_ASYNC_LDS 1
#else
#define USE_ASYNC_LDS 0
#endif

namespace {
constexpr int kB     = 4;
constexpr int kT     = 2048;
constexpr int kD     = 1024;
constexpr int kHeads = 16;
constexpr int kHD    = 64;
constexpr int kMH    = 4096;
constexpr int kM     = kB * kT;
constexpr int k6D    = 6 * kD;
}

__device__ __forceinline__ float sigmoidf_(float x) { return 1.f / (1.f + __expf(-x)); }
__device__ __forceinline__ float siluf_(float x)    { return x * sigmoidf_(x); }

__device__ __forceinline__ float block_sum(float v, float* red) {
  const int tid = threadIdx.x;
  red[tid] = v;
  __syncthreads();
  for (int s = blockDim.x >> 1; s > 0; s >>= 1) {
    if (tid < s) red[tid] += red[tid + s];
    __syncthreads();
  }
  float r = red[0];
  __syncthreads();
  return r;
}

__device__ __forceinline__ float block_max(float v, float* red) {
  const int tid = threadIdx.x;
  red[tid] = v;
  __syncthreads();
  for (int s = blockDim.x >> 1; s > 0; s >>= 1) {
    if (tid < s) red[tid] = fmaxf(red[tid], red[tid + s]);
    __syncthreads();
  }
  float r = red[0];
  __syncthreads();
  return r;
}

__device__ __forceinline__ void cp16(const int8_t* __restrict__ g, int8_t* l) {
#if USE_ASYNC_LDS
  __builtin_amdgcn_global_load_async_to_lds_b128(
      (__attribute__((address_space(1))) gv4i*)(void*)const_cast<int8_t*>(g),
      (__attribute__((address_space(3))) gv4i*)(void*)l, 0, 0);
#else
  *(int4*)l = *(const int4*)g;
#endif
}

__device__ __forceinline__ void async_drain() {
#if USE_ASYNC_LDS
  asm volatile("s_wait_asynccnt 0x0" ::: "memory");
#endif
}

__global__ __launch_bounds__(256)
void k_adaln(const float* __restrict__ c, const float* __restrict__ w,
             const float* __restrict__ bias, float* __restrict__ params) {
  int j = blockIdx.x * blockDim.x + threadIdx.x;
  if (j >= kB * k6D) return;
  int b   = j / k6D;
  int col = j % k6D;
  const float* cr = c + (size_t)b * kD;
  const float* wr = w + (size_t)col * kD;
  float acc = bias[col];
  for (int d = 0; d < kD; ++d) acc += siluf_(cr[d]) * wr[d];
  params[j] = acc;
}

__global__ __launch_bounds__(256)
void k_wabs_partial(const float* __restrict__ w, size_t n, float* __restrict__ partial) {
  __shared__ float red[256];
  float s = 0.f;
  for (size_t i = blockIdx.x * blockDim.x + threadIdx.x; i < n;
       i += (size_t)gridDim.x * blockDim.x)
    s += fabsf(w[i]);
  float t = block_sum(s, red);
  if (threadIdx.x == 0) partial[blockIdx.x] = t;
}

__global__ __launch_bounds__(256)
void k_wscale_finish(const float* __restrict__ partial, int nparts, float n_elems,
                     float* __restrict__ ws) {
  __shared__ float red[256];
  float s = 0.f;
  for (int i = threadIdx.x; i < nparts; i += 256) s += partial[i];
  float tot = block_sum(s, red);
  if (threadIdx.x == 0) {
    float mean = tot / n_elems;
    ws[0] = 1.f / fmaxf(mean, 1e-5f);
  }
}

__global__ __launch_bounds__(256)
void k_wquant(const float* __restrict__ w, const float* __restrict__ ws,
              int8_t* __restrict__ q, size_t n) {
  const float wsv = ws[0];
  for (size_t i = blockIdx.x * blockDim.x + threadIdx.x; i < n;
       i += (size_t)gridDim.x * blockDim.x) {
    float v = rintf(w[i] * wsv);
    v = fminf(1.f, fmaxf(-1.f, v));
    q[i] = (int8_t)v;
  }
}

__global__ __launch_bounds__(256)
void k_ln_mod_quant(const float* __restrict__ X, const float* __restrict__ params,
                    int shift_off, int scale_off,
                    int8_t* __restrict__ Q, float* __restrict__ S, int D_) {
  __shared__ float red[256];
  const int row = blockIdx.x;
  const int b   = row / kT;
  const int tid = threadIdx.x;
  const int npt = D_ / 256;
  const float* x   = X + (size_t)row * D_;
  const float* shv = params + (size_t)b * k6D + shift_off;
  const float* scv = params + (size_t)b * k6D + scale_off;

  float xv[8];
  float s0 = 0.f;
  for (int j = 0; j < npt; ++j) { float v = x[j * 256 + tid]; xv[j] = v; s0 += v; }
  const float mu = block_sum(s0, red) / (float)D_;

  float s1 = 0.f;
  for (int j = 0; j < npt; ++j) { float d = xv[j] - mu; s1 += d * d; }
  const float rstd = rsqrtf(block_sum(s1, red) / (float)D_ + 1e-6f);

  float s2 = 0.f;
  for (int j = 0; j < npt; ++j) {
    int d = j * 256 + tid;
    float y = (xv[j] - mu) * rstd * (1.f + scv[d]) + shv[d];
    xv[j] = y;
    s2 += y * y;
  }
  const float rn = rsqrtf(block_sum(s2, red) / (float)D_ + 1e-8f);

  float mx = 0.f;
  for (int j = 0; j < npt; ++j) { xv[j] *= rn; mx = fmaxf(mx, fabsf(xv[j])); }
  const float amax = block_max(mx, red);
  const float s = 127.f / fmaxf(amax, 1e-5f);

  for (int j = 0; j < npt; ++j) {
    float q = rintf(xv[j] * s);
    q = fminf(127.f, fmaxf(-128.f, q));
    Q[(size_t)row * D_ + j * 256 + tid] = (int8_t)q;
  }
  if (tid == 0) S[row] = s;
}

__global__ __launch_bounds__(256)
void k_rms_quant(const float* __restrict__ X, int8_t* __restrict__ Q,
                 float* __restrict__ S, int D_) {
  __shared__ float red[256];
  const int row = blockIdx.x;
  const int tid = threadIdx.x;
  const int npt = D_ / 256;
  const float* x = X + (size_t)row * D_;

  float xv[16];
  float s2 = 0.f;
  for (int j = 0; j < npt; ++j) { float v = x[j * 256 + tid]; xv[j] = v; s2 += v * v; }
  const float rn = rsqrtf(block_sum(s2, red) / (float)D_ + 1e-8f);

  float mx = 0.f;
  for (int j = 0; j < npt; ++j) { xv[j] *= rn; mx = fmaxf(mx, fabsf(xv[j])); }
  const float amax = block_max(mx, red);
  const float s = 127.f / fmaxf(amax, 1e-5f);

  for (int j = 0; j < npt; ++j) {
    float q = rintf(xv[j] * s);
    q = fminf(127.f, fmaxf(-128.f, q));
    Q[(size_t)row * D_ + j * 256 + tid] = (int8_t)q;
  }
  if (tid == 0) S[row] = s;
}

__global__ __launch_bounds__(256)
void k_gemm_i8(const int8_t* __restrict__ Aq, const float* __restrict__ a_s,
               const int8_t* __restrict__ Wq, const float* __restrict__ w_s,
               float* __restrict__ Cout, int M, int N, int K) {
  constexpr int TM = 256, TN = 64, TK = 64;
  __shared__ __align__(16) int8_t sA[2][TM * TK];
  __shared__ __align__(16) int8_t sB[2][TN * TK];

  const int tid  = threadIdx.x;
  const int lane = tid & 31;
  const int wave = tid >> 5;
  const int half = lane >> 4;
  const int l16  = lane & 15;
  const int row0 = blockIdx.y * TM;
  const int col0 = blockIdx.x * TN;

  v8i zero = {0, 0, 0, 0, 0, 0, 0, 0};
  v8i acc[8];
  #pragma unroll
  for (int i = 0; i < 8; ++i) acc[i] = zero;

  auto stage = [&](int buf, int k0) {
    #pragma unroll
    for (int p = 0; p < 4; ++p) {
      int cc = tid + p * 256;
      int r  = cc >> 2;
      int o  = (cc & 3) << 4;
      cp16(Aq + (size_t)(row0 + r) * K + k0 + o, &sA[buf][r * TK + o]);
    }
    int rb = tid >> 2;
    int ob = (tid & 3) << 4;
    cp16(Wq + (size_t)(col0 + rb) * K + k0 + ob, &sB[buf][rb * TK + ob]);
  };

  stage(0, 0);
  async_drain();
  __syncthreads();

  const int nk = K / TK;
  for (int it = 0; it < nk; ++it) {
    const int cur = it & 1;
    if (it + 1 < nk) stage(cur ^ 1, (it + 1) * TK);

    v8i a[2];
    #pragma unroll
    for (int f = 0; f < 2; ++f) {
      const int8_t* ar = &sA[cur][(wave * 32 + f * 16 + l16) * TK];
      #pragma unroll
      for (int p = 0; p < 4; ++p) {
        int2 d = *(const int2*)(ar + p * 16 + half * 8);
        a[f][2 * p]     = d.x;
        a[f][2 * p + 1] = d.y;
      }
    }

    #pragma unroll
    for (int t = 0; t < 4; ++t) {
      const int8_t* br = &sB[cur][(t * 16 + l16) * TK];
      v8i bf;
      #pragma unroll
      for (int p = 0; p < 4; ++p) {
        int2 d = *(const int2*)(br + p * 16 + half * 8);
        bf[2 * p]     = d.x;
        bf[2 * p + 1] = d.y;
      }
      acc[t] = __builtin_amdgcn_wmma_i32_16x16x64_iu8(
          true, a[0], true, bf, acc[t], false, false);
      acc[4 + t] = __builtin_amdgcn_wmma_i32_16x16x64_iu8(
          true, a[1], true, bf, acc[4 + t], false, false);
    }

    async_drain();
    __syncthreads();
  }

  const float ws = w_s[0];
  #pragma unroll
  for (int f = 0; f < 2; ++f) {
    #pragma unroll
    for (int t = 0; t < 4; ++t) {
      int n = col0 + t * 16 + l16;
      #pragma unroll
      for (int r = 0; r < 8; ++r) {
        int m = row0 + wave * 32 + f * 16 + half * 8 + r;
        Cout[(size_t)m * N + n] = (float)acc[f * 4 + t][r] / (a_s[m] * ws);
      }
    }
  }
}

__global__ __launch_bounds__(256)
void k_scan(float* __restrict__ i_raw, const float* __restrict__ f_raw) {
  int ch = blockIdx.x * blockDim.x + threadIdx.x;
  int b  = ch / kD;
  int d  = ch % kD;
  size_t base = (size_t)b * kT * kD + d;
  float h = 0.f;
  for (int t = 0; t < kT; ++t) {
    size_t idx = base + (size_t)t * kD;
    float f  = sigmoidf_(f_raw[idx]);
    float iv = siluf_(i_raw[idx]) * (1.f - f);
    h = f * h + iv;
    i_raw[idx] = h;
  }
}

__global__ __launch_bounds__(64)
void k_gatenorm(const float* __restrict__ h, const float* __restrict__ g,
                const float* __restrict__ gnw, float* __restrict__ o) {
  __shared__ float red[64];
  const int hd = threadIdx.x;
  const size_t base = (size_t)blockIdx.x * kHD;
  float hv = h[base + hd];
  red[hd] = hv * hv;
  __syncthreads();
  for (int s = 32; s > 0; s >>= 1) {
    if (hd < s) red[hd] += red[hd + s];
    __syncthreads();
  }
  float rms = rsqrtf(red[0] / (float)kHD + 1e-5f);
  float gv  = g[base + hd];
  o[base + hd] = hv * rms * gnw[hd] * (gv * sigmoidf_(gv));
}

__global__ __launch_bounds__(256)
void k_resgate(const float* __restrict__ xin, const float* __restrict__ y,
               const float* __restrict__ params, int gate_off,
               float* __restrict__ xout) {
  const size_t total = (size_t)kM * kD;
  for (size_t i = blockIdx.x * blockDim.x + threadIdx.x; i < total;
       i += (size_t)gridDim.x * blockDim.x) {
    int b = (int)(i / ((size_t)kT * kD));
    int d = (int)(i % kD);
    xout[i] = xin[i] + params[(size_t)b * k6D + gate_off + d] * y[i];
  }
}

__global__ __launch_bounds__(256)
void k_swiglu(const float* __restrict__ y, float* __restrict__ mid) {
  const size_t total = (size_t)kM * kMH;
  for (size_t i = blockIdx.x * blockDim.x + threadIdx.x; i < total;
       i += (size_t)gridDim.x * blockDim.x) {
    size_t r  = i / kMH;
    int    cc = (int)(i % kMH);
    float gvar = y[r * (2 * kMH) + cc];
    float vv   = y[r * (2 * kMH) + kMH + cc];
    mid[i] = siluf_(gvar) * vv;
  }
}

extern "C" void kernel_launch(void* const* d_in, const int* in_sizes, int n_in,
                              void* d_out, int out_size, void* d_ws, size_t ws_size,
                              hipStream_t stream) {
  (void)in_sizes; (void)n_in; (void)out_size; (void)ws_size;

  const float* x       = (const float*)d_in[0];
  const float* c       = (const float*)d_in[1];
  const float* adaln_w = (const float*)d_in[2];
  const float* adaln_b = (const float*)d_in[3];
  const float* wi      = (const float*)d_in[4];
  const float* wf      = (const float*)d_in[5];
  const float* wg      = (const float*)d_in[6];
  const float* gnorm_w = (const float*)d_in[7];
  const float* wo      = (const float*)d_in[8];
  const float* gate_w  = (const float*)d_in[9];
  const float* down_w  = (const float*)d_in[10];
  float* out = (float*)d_out;

  uint8_t* base = (uint8_t*)d_ws;
  size_t off = 0;
  auto alloc = [&](size_t bytes) -> void* {
    void* p = base + off;
    off = (off + bytes + 255) & ~(size_t)255;
    return p;
  };
  float*  params  = (float*)alloc((size_t)kB * k6D * sizeof(float));
  float*  wsc     = (float*)alloc(8 * sizeof(float));
  float*  wpart   = (float*)alloc(1024 * sizeof(float));
  int8_t* q_wi    = (int8_t*)alloc((size_t)kD * kD);
  int8_t* q_wf    = (int8_t*)alloc((size_t)kD * kD);
  int8_t* q_wg    = (int8_t*)alloc((size_t)kD * kD);
  int8_t* q_wo    = (int8_t*)alloc((size_t)kD * kD);
  int8_t* q_gate  = (int8_t*)alloc((size_t)2 * kMH * kD);
  int8_t* q_down  = (int8_t*)alloc((size_t)kD * kMH);
  int8_t* actq    = (int8_t*)alloc((size_t)kM * kMH);
  float*  act_s   = (float*)alloc((size_t)kM * sizeof(float));
  float*  iraw    = (float*)alloc((size_t)kM * kD * sizeof(float));
  float*  fraw    = (float*)alloc((size_t)kM * kD * sizeof(float));
  float*  graw    = (float*)alloc((size_t)kM * kD * sizeof(float));
  float*  x1      = (float*)alloc((size_t)kM * kD * sizeof(float));
  float*  bigN    = (float*)alloc((size_t)kM * 2 * kMH * sizeof(float));
  float*  mid     = (float*)alloc((size_t)kM * kMH * sizeof(float));

  k_adaln<<<(kB * k6D + 255) / 256, 256, 0, stream>>>(c, adaln_w, adaln_b, params);

  struct WSpec { const float* w; size_t n; int8_t* q; int idx; };
  WSpec specs[6] = {
      {wi,     (size_t)kD * kD,      q_wi,   0},
      {wf,     (size_t)kD * kD,      q_wf,   1},
      {wg,     (size_t)kD * kD,      q_wg,   2},
      {wo,     (size_t)kD * kD,      q_wo,   3},
      {gate_w, (size_t)2 * kMH * kD, q_gate, 4},
      {down_w, (size_t)kD * kMH,     q_down, 5},
  };
  for (int i = 0; i < 6; ++i) {
    k_wabs_partial<<<1024, 256, 0, stream>>>(specs[i].w, specs[i].n, wpart);
    k_wscale_finish<<<1, 256, 0, stream>>>(wpart, 1024, (float)specs[i].n,
                                           wsc + specs[i].idx);
    k_wquant<<<2048, 256, 0, stream>>>(specs[i].w, wsc + specs[i].idx,
                                       specs[i].q, specs[i].n);
  }

  auto gemm = [&](const int8_t* A, const float* As, const int8_t* W,
                  const float* Ws, float* C_, int M_, int N_, int K_) {
    dim3 g(N_ / 64, M_ / 256);
    k_gemm_i8<<<g, 256, 0, stream>>>(A, As, W, Ws, C_, M_, N_, K_);
  };

  k_ln_mod_quant<<<kM, 256, 0, stream>>>(x, params,   0,   kD,
                                         actq, act_s, kD);
  gemm(actq, act_s, q_wi, wsc + 0, iraw, kM, kD, kD);
  gemm(actq, act_s, q_wf, wsc + 1, fraw, kM, kD, kD);
  gemm(actq, act_s, q_wg, wsc + 2, graw, kM, kD, kD);

  k_scan<<<(kB * kD) / 256, 256, 0, stream>>>(iraw, fraw);
  k_gatenorm<<<kB * kT * kHeads, 64, 0, stream>>>(iraw, graw, gnorm_w, fraw);

  k_rms_quant<<<kM, 256, 0, stream>>>(fraw, actq, act_s, kD);
  gemm(actq, act_s, q_wo, wsc + 3, graw, kM, kD, kD);
  k_resgate<<<4096, 256, 0, stream>>>(x, graw, params,   2 * kD, x1);

  k_ln_mod_quant<<<kM, 256, 0, stream>>>(x1, params,   3 * kD,
                                           4 * kD, actq, act_s, kD);
  gemm(actq, act_s, q_gate, wsc + 4, bigN, kM, 2 * kMH, kD);
  k_swiglu<<<8192, 256, 0, stream>>>(bigN, mid);
  k_rms_quant<<<kM, 256, 0, stream>>>(mid, actq, act_s, kMH);
  gemm(actq, act_s, q_down, wsc + 5, graw, kM, kD, kMH);
  k_resgate<<<4096, 256, 0, stream>>>(x1, graw, params,   5 * kD, out);
}
